// alinet_atten_77163382440890
// MI455X (gfx1250) — matrix operand order repaired and checked by reading, not compiled, not run on hardware
//
#include <hip/hip_runtime.h>
#include <math.h>

typedef __attribute__((ext_vector_type(16))) _Float16 v16h;
typedef __attribute__((ext_vector_type(8)))  _Float16 v8h;
typedef __attribute__((ext_vector_type(8)))  float    v8f;

#define FDIM 128
#define BN_EPS 1e-3f
#define LRELU 0.2f

// ---------------------------------------------------------------------------
// Pack f32 [128x128] weight matrices into f16 WMMA B fragments in contiguous k order (A uses the same).
// Per matrix: tiles (kt=0..3 over K, nt=0..7 over N); each tile = 32 lanes x
// 16 halves (1 KB).  B is 32x16 (KxN): lanes 0-15 hold K=0..15 of column
// (nt*16+lane), lanes 16-31 hold K=16..31 of column (nt*16+lane-16).
// ---------------------------------------------------------------------------
__global__ void prep_panels(const float* __restrict__ W,
                            const float* __restrict__ M1,
                            const float* __restrict__ M2,
                            _Float16* __restrict__ panels) {
    int idx = blockIdx.x * blockDim.x + threadIdx.x;
    const int per = 4 * 8 * 512;               // 16384 halves per matrix
    if (idx >= 3 * per) return;
    int mat  = idx / per;
    int r    = idx % per;
    int tile = r >> 9;
    int rem  = r & 511;
    int lane = rem >> 4;
    int h    = rem & 15;
    int kt   = tile >> 3;
    int nt   = tile & 7;
    int k    = kt * 32 + ((lane >= 16) ? 16 : 0) + h;
    int nn   = nt * 16 + (lane & 15);
    const float* src = (mat == 0) ? W : ((mat == 1) ? M1 : M2);
    panels[idx] = (_Float16)src[k * FDIM + nn];
}

// 16x16x128 f16->f32 tile: 8 global_load_b128 (one clause) + 4 chained WMMAs
__device__ __forceinline__ v8f wmma_tile128(const v16h* afrag,
                                            const _Float16* __restrict__ pb,
                                            int nt, int lane) {
    v16h bf[4];
#pragma unroll
    for (int kt = 0; kt < 4; ++kt)
        bf[kt] = *(const v16h*)(pb + (kt * 8 + nt) * 512 + lane * 16);
    v8f acc = {};
#pragma unroll
    for (int kt = 0; kt < 4; ++kt)
        acc = __builtin_amdgcn_wmma_f32_16x16x32_f16(
                false, afrag[kt], false, bf[kt], (short)0, acc, false, false);
    return acc;
}

// ---------------------------------------------------------------------------
// Fused: BatchNorm -> xn(f16, LDS) -> WMMA GEMMs vs W/M1/M2.
// Block = 256 threads = 8 waves; each wave owns a 16-row tile (block: 128 rows).
//   W:     store mapped[row, :]  (f32, npad-row padded -> unguarded stores)
//   M1/M2: s = tanh( sum_n (xn@M)[r,n] * xn[r,n] )
// A-fragments (xn) are loaded once and reused for all 3 matrices x 8 N-tiles.
// mapped/s1/s2 are padded to a multiple of 128 rows, so no store guards.
// ---------------------------------------------------------------------------
__global__ __launch_bounds__(256)
void bn_gemm_scores(const float* __restrict__ x,
                    const float* __restrict__ gamma,
                    const float* __restrict__ beta,
                    const float* __restrict__ bn_mean,
                    const float* __restrict__ bn_var,
                    const _Float16* __restrict__ panels,
                    float* __restrict__ mapped,
                    float* __restrict__ s1,
                    float* __restrict__ s2,
                    int n) {
    __shared__ _Float16 xh[128 * 136];          // 16-byte aligned row stride
    __shared__ float csc[128], csh[128];

    int tid = threadIdx.x;
    if (tid < 128) {
        float inv = rsqrtf(bn_var[tid] + BN_EPS);
        float sc  = gamma[tid] * inv;
        csc[tid]  = sc;
        csh[tid]  = beta[tid] - bn_mean[tid] * sc;
    }
    __syncthreads();

    int gRow0 = blockIdx.x * 128;
    for (int t = tid; t < 128 * 128; t += 256) {
        int r = t >> 7, c = t & 127;
        int gr = gRow0 + r;
        float v = 0.f;
        if (gr < n) v = x[(size_t)gr * FDIM + c] * csc[c] + csh[c];
        xh[r * 136 + c] = (_Float16)v;
    }
    __syncthreads();

    int lane = tid & 31, wave = tid >> 5;
    int lq = lane & 15, hi = lane >> 4;

    // pull weight panels toward the WGP cache (global_prefetch_b8)
    __builtin_prefetch(panels + lane * 16, 0, 3);

    // A fragments: 16x32 per kt, contiguous k order (same as B panels).
    // lane<16: row=lq, K = kt*32 + {0..15}; lane>=16: K = kt*32 + {16..31}
    int arow = wave * 16 + lq;
    v16h afrag[4];
#pragma unroll
    for (int kt = 0; kt < 4; ++kt) {
        int koff = kt * 32 + hi * 16;
        v8h lo = *(const v8h*)&xh[arow * 136 + koff];
        v8h hp = *(const v8h*)&xh[arow * 136 + koff + 8];
        v16h a;
#pragma unroll
        for (int i = 0; i < 8; ++i) { a[i] = lo[i]; a[i + 8] = hp[i]; }
        afrag[kt] = a;
    }

    int tileRow0 = gRow0 + wave * 16;
    const _Float16* pbW  = panels;
    const _Float16* pbM1 = panels + 16384;
    const _Float16* pbM2 = panels + 32768;

    float sacc1[8], sacc2[8];
#pragma unroll
    for (int v = 0; v < 8; ++v) { sacc1[v] = 0.f; sacc2[v] = 0.f; }

#pragma unroll
    for (int nt = 0; nt < 8; ++nt) {
        // ---- W tile: mapped[tileRow0 + v + hi*8, nt*16+lq] = acc[v] ----
        {
            v8f acc = wmma_tile128(afrag, pbW, nt, lane);
            float* mrow = mapped + (size_t)tileRow0 * FDIM + nt * 16 + lq;
#pragma unroll
            for (int v = 0; v < 8; ++v)
                mrow[(size_t)(v + hi * 8) * FDIM] = acc[v];
        }

        // xn values for this nt's C-layout cells, shared by M1 and M2 paths
        float xv[8];
#pragma unroll
        for (int v = 0; v < 8; ++v) {
            int lr = wave * 16 + v + hi * 8;
            xv[v] = (float)xh[lr * 136 + nt * 16 + lq];
        }

        // ---- M1 tile: quadratic-form partials ----
        {
            v8f acc = wmma_tile128(afrag, pbM1, nt, lane);
#pragma unroll
            for (int v = 0; v < 8; ++v) sacc1[v] += acc[v] * xv[v];
        }

        // ---- M2 tile ----
        {
            v8f acc = wmma_tile128(afrag, pbM2, nt, lane);
#pragma unroll
            for (int v = 0; v < 8; ++v) sacc2[v] += acc[v] * xv[v];
        }
    }

    // reduce quadratic-form partials across each 16-lane half, write tanh
#pragma unroll
    for (int v = 0; v < 8; ++v) {
        float p1 = sacc1[v], p2 = sacc2[v];
#pragma unroll
        for (int off = 1; off < 16; off <<= 1) {
            p1 += __shfl_xor(p1, off, 32);
            p2 += __shfl_xor(p2, off, 32);
        }
        int gr = tileRow0 + v + hi * 8;
        if (lq == 0) {                  // rows padded: no n-guard needed
            s1[gr] = tanhf(p1);
            s2[gr] = tanhf(p2);
        }
    }
}

// ---------------------------------------------------------------------------
// init: out = 0, segment max = -inf, denom = 0
// ---------------------------------------------------------------------------
__global__ void init_out(float* __restrict__ out, float* __restrict__ m,
                         float* __restrict__ denom, int n) {
    int i = blockIdx.x * blockDim.x + threadIdx.x;
    if (i < n * FDIM) out[i] = 0.f;
    if (i < n) { m[i] = -INFINITY; denom[i] = 0.f; }
}

__device__ __forceinline__ void atomicMaxF(float* addr, float val) {
    if (__float_as_int(val) >= 0)
        atomicMax((int*)addr, __float_as_int(val));
    else
        atomicMin((unsigned int*)addr, __float_as_uint(val));
}

// e = leakyrelu(a*(s1[row] + s2[col]));  segment max over row
__global__ void edge_max(const float* __restrict__ a,
                         const float* __restrict__ s1,
                         const float* __restrict__ s2,
                         const int* __restrict__ row,
                         const int* __restrict__ col,
                         float* __restrict__ ebuf,
                         float* __restrict__ m, int e) {
    int i = blockIdx.x * blockDim.x + threadIdx.x;
    if (i >= e) return;
    int r = row[i], c = col[i];
    float v = a[i] * s1[r] + a[i] * s2[c];
    v = (v >= 0.f) ? v : LRELU * v;
    ebuf[i] = v;
    atomicMaxF(&m[r], v);
}

// p = exp(e - m[row]);  denom[row] += p
__global__ void edge_expsum(const int* __restrict__ row,
                            const float* __restrict__ m,
                            float* __restrict__ ebuf,
                            float* __restrict__ denom, int e) {
    int i = blockIdx.x * blockDim.x + threadIdx.x;
    if (i >= e) return;
    int r = row[i];
    float p = expf(ebuf[i] - m[r]);
    ebuf[i] = p;
    atomicAdd(&denom[r], p);
}

// out[row] += (p/denom[row]) * mapped[col] ; one wave32 per edge, 4 floats/lane
__global__ void edge_aggregate(const int* __restrict__ row,
                               const int* __restrict__ col,
                               const float* __restrict__ ebuf,
                               const float* __restrict__ denom,
                               const float* __restrict__ mapped,
                               float* __restrict__ out, int e) {
    int gt = blockIdx.x * blockDim.x + threadIdx.x;
    int w = gt >> 5, lane = gt & 31;
    if (w >= e) return;
    int r = row[w], c = col[w];
    float alpha = ebuf[w] / denom[r];
    const float4* mp = (const float4*)(mapped + (size_t)c * FDIM);
    float4 v = mp[lane];
    float* o = out + (size_t)r * FDIM + lane * 4;
    atomicAdd(o + 0, alpha * v.x);
    atomicAdd(o + 1, alpha * v.y);
    atomicAdd(o + 2, alpha * v.z);
    atomicAdd(o + 3, alpha * v.w);
}

// ---------------------------------------------------------------------------
extern "C" void kernel_launch(void* const* d_in, const int* in_sizes, int n_in,
                              void* d_out, int out_size, void* d_ws, size_t ws_size,
                              hipStream_t stream) {
    const float* x       = (const float*)d_in[0];
    const float* a_vals  = (const float*)d_in[1];
    const float* W       = (const float*)d_in[2];
    const float* M1      = (const float*)d_in[3];
    const float* M2      = (const float*)d_in[4];
    const float* gamma   = (const float*)d_in[5];
    const float* beta    = (const float*)d_in[6];
    const float* bn_mean = (const float*)d_in[7];
    const float* bn_var  = (const float*)d_in[8];
    const int*   row     = (const int*)d_in[9];
    const int*   col     = (const int*)d_in[10];

    int n = in_sizes[0] / FDIM;
    int e = in_sizes[9];
    int npad = (n + 127) & ~127;       // padded rows -> unguarded WMMA stores
    float* out = (float*)d_out;

    // workspace carve-out
    char* ws = (char*)d_ws;
    size_t off = 0;
    auto carve = [&](size_t bytes) -> void* {
        void* p = ws + off;
        off = (off + bytes + 255) & ~(size_t)255;
        return p;
    };
    _Float16* panels = (_Float16*)carve(3 * 16384 * sizeof(_Float16));
    float* mapped = (float*)carve((size_t)npad * FDIM * sizeof(float));
    float* s1     = (float*)carve((size_t)npad * sizeof(float));
    float* s2     = (float*)carve((size_t)npad * sizeof(float));
    float* m      = (float*)carve((size_t)n * sizeof(float));
    float* denom  = (float*)carve((size_t)n * sizeof(float));
    float* ebuf   = (float*)carve((size_t)e * sizeof(float));

    hipLaunchKernelGGL(prep_panels, dim3((3 * 16384 + 255) / 256), dim3(256), 0, stream,
                       W, M1, M2, panels);
    hipLaunchKernelGGL(bn_gemm_scores, dim3(npad / 128), dim3(256), 0, stream,
                       x, gamma, beta, bn_mean, bn_var, panels, mapped, s1, s2, n);
    long long nf = (long long)n * FDIM;
    hipLaunchKernelGGL(init_out, dim3((unsigned)((nf + 255) / 256)), dim3(256), 0, stream,
                       out, m, denom, n);
    hipLaunchKernelGGL(edge_max, dim3((e + 255) / 256), dim3(256), 0, stream,
                       a_vals, s1, s2, row, col, ebuf, m, e);
    hipLaunchKernelGGL(edge_expsum, dim3((e + 255) / 256), dim3(256), 0, stream,
                       row, m, ebuf, denom, e);
    long long et = (long long)e * 32;
    hipLaunchKernelGGL(edge_aggregate, dim3((unsigned)((et + 255) / 256)), dim3(256), 0, stream,
                       row, col, ebuf, denom, mapped, out, e);
}
